// SlidingWindowAttention_30039001268738
// MI455X (gfx1250) — compile-verified, matrix operand order checked by reading, not run on hardware
//
#include <hip/hip_runtime.h>
#include <hip/hip_bf16.h>
#include <math.h>

// ---------------------------------------------------------------------------
// Problem constants (match reference)
// ---------------------------------------------------------------------------
constexpr int B_   = 2;
constexpr int T_   = 2048;
constexpr int DM   = 2048;   // d_model
constexpr int NH   = 16;     // query heads
constexpr int NKV  = 4;      // kv heads
constexpr int HD   = 128;    // head dim
constexpr int NREP = NH / NKV;

typedef __attribute__((ext_vector_type(16))) __bf16 v16bf;
typedef __attribute__((ext_vector_type(8)))  __bf16 v8bf;
typedef __attribute__((ext_vector_type(4)))  __bf16 v4bf;
typedef __attribute__((ext_vector_type(8)))  float  v8f;
typedef int v4i __attribute__((vector_size(16)));   // matches builtin param type

// CDNA5 async global->LDS copy (GLOBAL_LOAD_ASYNC_TO_LDS_B128, ASYNCcnt).
// Probe-derived signature: (v4i __device__* src, AS3 dst, imm offset, imm cpol).
#if defined(__has_builtin)
#if __has_builtin(__builtin_amdgcn_global_load_async_to_lds_b128)
#define HAVE_ASYNC_LDS 1
#endif
#endif

#ifdef HAVE_ASYNC_LDS
#define ASYNC_CP16(gsrc, ldst)                                                 \
  __builtin_amdgcn_global_load_async_to_lds_b128(                              \
      (__attribute__((address_space(1))) v4i*)(                                \
          (__attribute__((address_space(1))) void*)(void*)(gsrc)),             \
      (__attribute__((address_space(3))) v4i*)(                                \
          (__attribute__((address_space(3))) void*)(void*)(ldst)),             \
      0, 0)
#define ASYNC_DRAIN() asm volatile("s_wait_asynccnt 0" ::: "memory")
#else
#define ASYNC_CP16(gsrc, ldst) (*(v8bf*)(ldst) = *(const v8bf*)(gsrc))
#define ASYNC_DRAIN() ((void)0)
#endif

// D = A(16x32 bf16) * B(32x16 bf16) + C(16x16 f32)
__device__ inline v8f wmma_bf16(v16bf a, v16bf b, v8f c) {
  return __builtin_amdgcn_wmma_f32_16x16x32_bf16(
      /*neg_a=*/false, a, /*neg_b=*/false, b,
      /*c_mod=*/(short)0, c, /*reuse_a=*/false, /*reuse_b=*/false);
}

// Load one 16x32 (or 32x16) bf16 fragment from a [rows][ld] row-major LDS/global
// region. Caller passes p = &base[(rowBase + lane%16)*ld + colBase + (lane/16)*8].
// Per ISA layout: lanes 0-15 hold K=0..7 (elems 0-7) and K=16..23 (elems 8-15);
// lanes 16-31 hold K=8..15 and K=24..31.  Two 16B loads.
__device__ inline v16bf ld_frag(const __bf16* p) {
  v16bf f;
#pragma unroll
  for (int i = 0; i < 8; ++i) { f[i] = p[i]; f[i + 8] = p[i + 16]; }
  return f;
}

// ---------------------------------------------------------------------------
// fp32 -> bf16 convert (memory bound)
// ---------------------------------------------------------------------------
__global__ __launch_bounds__(256)
void cvt_f32_bf16(const float* __restrict__ in, __bf16* __restrict__ out, long n) {
  long i = ((long)blockIdx.x * blockDim.x + threadIdx.x) * 4;
  if (i >= n) return;
  float4 f = *(const float4*)(in + i);
  v4bf o = { (__bf16)f.x, (__bf16)f.y, (__bf16)f.z, (__bf16)f.w };
  *(v4bf*)(out + i) = o;
}

// ---------------------------------------------------------------------------
// RMS norm over rows of 128 + scale, fp32 in -> bf16 out. One wave per row.
// ---------------------------------------------------------------------------
__global__ __launch_bounds__(256)
void qknorm_kernel(const float* __restrict__ in, const float* __restrict__ scale,
                   __bf16* __restrict__ out, int nrows) {
  int row  = blockIdx.x * 8 + (threadIdx.x >> 5);
  if (row >= nrows) return;
  int lane = threadIdx.x & 31;
  const float* p = in + (size_t)row * HD + lane * 4;
  float4 v  = *(const float4*)p;
  float ss  = v.x * v.x + v.y * v.y + v.z * v.z + v.w * v.w;
#pragma unroll
  for (int m = 16; m >= 1; m >>= 1) ss += __shfl_xor(ss, m, 32);
  float r = rsqrtf(ss * (1.0f / HD) + 1e-6f);
  float4 sc = *(const float4*)(scale + lane * 4);
  v4bf o = { (__bf16)(v.x * r * sc.x), (__bf16)(v.y * r * sc.y),
             (__bf16)(v.z * r * sc.z), (__bf16)(v.w * r * sc.w) };
  *(v4bf*)(out + (size_t)row * HD + lane * 4) = o;
}

// ---------------------------------------------------------------------------
// WMMA GEMM:  C[m,n] = sum_k A[m*K+k] * Bm[n*K+k]   (A: MxK, Bm: NxK, C fp32)
// Block tile 128x128, BK=32, 256 threads = 8 waves; each wave owns 64x32
// (4 M-subtiles x 2 N-subtiles of 16x16). Tiles staged via async global->LDS.
// ---------------------------------------------------------------------------
constexpr int LDA = 40;   // 32 + 8 pad (bf16 elems)

__global__ __launch_bounds__(256)
void gemm_bf16_nt(const __bf16* __restrict__ A, const __bf16* __restrict__ Bm,
                  float* __restrict__ C, int M, int N, int K) {
  __shared__ __bf16 sA[128 * LDA];
  __shared__ __bf16 sB[128 * LDA];

  const int tid  = threadIdx.x;
  const int lane = tid & 31, wid = tid >> 5;
  const int lr   = lane & 15, lh  = lane >> 4;
  const int mb   = M / 128;
  const int bm   = (blockIdx.x % mb) * 128;
  const int bn   = (blockIdx.x / mb) * 128;
  const int wm   = (wid >> 2) * 64;       // 0 / 64
  const int wn   = (wid & 3) * 32;        // 0 / 32 / 64 / 96

  v8f acc[4][2];
#pragma unroll
  for (int i = 0; i < 4; ++i)
#pragma unroll
    for (int j = 0; j < 2; ++j)
      acc[i][j] = (v8f){0.f,0.f,0.f,0.f,0.f,0.f,0.f,0.f};

  // Staging coordinates for this thread (2 chunks of 8 bf16 per matrix).
  const int r0 = tid >> 2,        c0 = (tid & 3) * 8;          // chunk 0
  const int r1 = (tid + 256) >> 2, c1 = ((tid + 256) & 3) * 8; // chunk 1

  for (int k0 = 0; k0 < K; k0 += 32) {
    ASYNC_CP16(&A[(size_t)(bm + r0) * K + k0 + c0], &sA[r0 * LDA + c0]);
    ASYNC_CP16(&A[(size_t)(bm + r1) * K + k0 + c1], &sA[r1 * LDA + c1]);
    ASYNC_CP16(&Bm[(size_t)(bn + r0) * K + k0 + c0], &sB[r0 * LDA + c0]);
    ASYNC_CP16(&Bm[(size_t)(bn + r1) * K + k0 + c1], &sB[r1 * LDA + c1]);
    ASYNC_DRAIN();
    __syncthreads();

    // Pull the next K-slab toward the WGP while WMMAs run.
    if (k0 + 32 < K) {
      __builtin_prefetch(&A[(size_t)(bm + r0) * K + k0 + 32 + c0], 0, 1);
      __builtin_prefetch(&Bm[(size_t)(bn + r0) * K + k0 + 32 + c0], 0, 1);
    }

    v16bf af[4], bf[2];
#pragma unroll
    for (int i = 0; i < 4; ++i)
      af[i] = ld_frag(&sA[(wm + i * 16 + lr) * LDA + lh * 8]);
#pragma unroll
    for (int j = 0; j < 2; ++j)
      bf[j] = ld_frag(&sB[(wn + j * 16 + lr) * LDA + lh * 8]);
#pragma unroll
    for (int i = 0; i < 4; ++i)
#pragma unroll
      for (int j = 0; j < 2; ++j)
        acc[i][j] = wmma_bf16(af[i], bf[j], acc[i][j]);
    __syncthreads();
  }

  // Store: C/D layout -> lane lr = N, VGPR v = M (v + 8*lh)
#pragma unroll
  for (int i = 0; i < 4; ++i)
#pragma unroll
    for (int j = 0; j < 2; ++j) {
      int row0 = bm + wm + i * 16 + lh * 8;
      int col  = bn + wn + j * 16 + lr;
#pragma unroll
      for (int v = 0; v < 8; ++v)
        C[(size_t)(row0 + v) * N + col] = acc[i][j][v];
    }
}

// ---------------------------------------------------------------------------
// Flash attention with causal mask + GQA (online softmax), bf16 WMMA.
// grid: B * NH * (T/128) blocks of 256 threads (8 waves x 16 query rows).
// Q/K/V are bf16 laid out as (b, t, head, d) row-major.
// Output: bf16 attn_out laid out (b*T + t, h*128 + d) = (4096 x 2048).
// ---------------------------------------------------------------------------
constexpr int LDK = 136;  // 128 + 8
constexpr int LDV = 72;   // 64 + 8
constexpr int LDP = 72;

__global__ __launch_bounds__(256)
void flash_attn_kernel(const __bf16* __restrict__ Q, const __bf16* __restrict__ Kk,
                       const __bf16* __restrict__ Vv, __bf16* __restrict__ O) {
  __shared__ __bf16 sK[64 * LDK];        // [key][dim]
  __shared__ __bf16 sVt[128 * LDV];      // [dim][key]  (transposed V)
  __shared__ __bf16 sP[8 * 16 * LDP];    // per-wave P patch [16 rows][64 keys]

  const int tiles = T_ / 128;
  const int b   = blockIdx.x / (NH * tiles);
  const int h   = (blockIdx.x / tiles) % NH;
  const int qt  = blockIdx.x % tiles;
  const int kvh = h / NREP;

  const int tid  = threadIdx.x;
  const int lane = tid & 31, wid = tid >> 5;
  const int lr   = lane & 15, lh  = lane >> 4;
  const int qrow0 = qt * 128 + wid * 16;

  // Load this wave's Q fragments: 16 rows x 128 dims = 4 chunks of 16x32.
  v16bf qf[4];
  {
    const __bf16* qp = Q + ((size_t)(b * T_ + qrow0 + lr) * NH + h) * HD;
#pragma unroll
    for (int kc = 0; kc < 4; ++kc)
      qf[kc] = ld_frag(qp + kc * 32 + lh * 8);
  }

  v8f oacc[8];
#pragma unroll
  for (int d = 0; d < 8; ++d) oacc[d] = (v8f){0.f,0.f,0.f,0.f,0.f,0.f,0.f,0.f};
  float mprev[8], lsum[8];
#pragma unroll
  for (int v = 0; v < 8; ++v) { mprev[v] = -1e30f; lsum[v] = 0.f; }

  const float sscale = 0.08838834764831845f;  // 1/sqrt(128)
  const int   nkb    = qt * 2 + 2;            // causal: 64-key blocks needed

  for (int kb = 0; kb < nkb; ++kb) {
    const int kbase = kb * 64;
    __syncthreads();  // protect LDS reuse across iterations
    // Stage K [64][128] (async, no VGPR round-trip) and V transposed [128][64]
    // (through VGPRs, since we transpose on the way in).
#pragma unroll
    for (int p = 0; p < 4; ++p) {
      int c    = tid + p * 256;     // 0..1023
      int kr   = c >> 4;            // key 0..63
      int kcol = (c & 15) * 8;      // dim chunk
      size_t gro = ((size_t)(b * T_ + kbase + kr) * NKV + kvh) * HD + kcol;
      ASYNC_CP16(Kk + gro, &sK[kr * LDK + kcol]);
      v8bf vv = *(const v8bf*)(Vv + gro);
#pragma unroll
      for (int e = 0; e < 8; ++e) sVt[(kcol + e) * LDV + kr] = vv[e];
    }
    ASYNC_DRAIN();
    __syncthreads();

    // S = Q * K^T : 16x64 in 4 subtiles of 16x16
    v8f s[4];
#pragma unroll
    for (int ns = 0; ns < 4; ++ns) {
      v8f a = (v8f){0.f,0.f,0.f,0.f,0.f,0.f,0.f,0.f};
#pragma unroll
      for (int kc = 0; kc < 4; ++kc) {
        v16bf kf = ld_frag(&sK[(ns * 16 + lr) * LDK + kc * 32 + lh * 8]);
        a = wmma_bf16(qf[kc], kf, a);
      }
      s[ns] = a;
    }

    // Scale + causal mask, row max
    float rmax[8];
#pragma unroll
    for (int v = 0; v < 8; ++v) rmax[v] = -1e30f;
#pragma unroll
    for (int ns = 0; ns < 4; ++ns) {
      const int kj = kbase + ns * 16 + lr;
#pragma unroll
      for (int v = 0; v < 8; ++v) {
        const int qi = qrow0 + v + 8 * lh;
        float x = s[ns][v] * sscale;
        if (kj > qi) x = -1e30f;
        s[ns][v] = x;
        rmax[v]  = fmaxf(rmax[v], x);
      }
    }
#pragma unroll
    for (int v = 0; v < 8; ++v) {
#pragma unroll
      for (int m = 1; m < 16; m <<= 1)
        rmax[v] = fmaxf(rmax[v], __shfl_xor(rmax[v], m, 32));
    }

    float corr[8], rsum[8];
#pragma unroll
    for (int v = 0; v < 8; ++v) {
      float mnew = fmaxf(mprev[v], rmax[v]);
      corr[v]  = __expf(mprev[v] - mnew);
      mprev[v] = mnew;
      rsum[v]  = 0.f;
    }
#pragma unroll
    for (int ns = 0; ns < 4; ++ns)
#pragma unroll
      for (int v = 0; v < 8; ++v) {
        float pv = __expf(s[ns][v] - mprev[v]);
        s[ns][v] = pv;
        rsum[v] += pv;
      }
#pragma unroll
    for (int v = 0; v < 8; ++v) {
#pragma unroll
      for (int m = 1; m < 16; m <<= 1)
        rsum[v] += __shfl_xor(rsum[v], m, 32);
      lsum[v] = lsum[v] * corr[v] + rsum[v];
    }
#pragma unroll
    for (int d = 0; d < 8; ++d)
#pragma unroll
      for (int v = 0; v < 8; ++v) oacc[d][v] *= corr[v];

    // C-layout -> A-layout for P via per-wave LDS patch
    __bf16* pw = &sP[wid * 16 * LDP];
#pragma unroll
    for (int ns = 0; ns < 4; ++ns)
#pragma unroll
      for (int v = 0; v < 8; ++v)
        pw[(v + 8 * lh) * LDP + ns * 16 + lr] = (__bf16)s[ns][v];

    v16bf pf[2];
#pragma unroll
    for (int kc2 = 0; kc2 < 2; ++kc2)
      pf[kc2] = ld_frag(&pw[lr * LDP + kc2 * 32 + lh * 8]);

    // O += P * V
#pragma unroll
    for (int d = 0; d < 8; ++d) {
#pragma unroll
      for (int kc2 = 0; kc2 < 2; ++kc2) {
        v16bf vf = ld_frag(&sVt[(d * 16 + lr) * LDV + kc2 * 32 + lh * 8]);
        oacc[d] = wmma_bf16(pf[kc2], vf, oacc[d]);
      }
    }
  }

  // Epilogue: O /= l, store bf16 at (b*T + qi, h*128 + dim)
  float inv[8];
#pragma unroll
  for (int v = 0; v < 8; ++v) inv[v] = 1.0f / lsum[v];
#pragma unroll
  for (int d = 0; d < 8; ++d) {
    int col = h * HD + d * 16 + lr;
#pragma unroll
    for (int v = 0; v < 8; ++v) {
      int qi = qrow0 + v + 8 * lh;
      O[(size_t)(b * T_ + qi) * DM + col] = (__bf16)(oacc[d][v] * inv[v]);
    }
  }
}

// ---------------------------------------------------------------------------
// Launch
// ---------------------------------------------------------------------------
extern "C" void kernel_launch(void* const* d_in, const int* in_sizes, int n_in,
                              void* d_out, int out_size, void* d_ws, size_t ws_size,
                              hipStream_t stream) {
  const float* x  = (const float*)d_in[0];
  const float* Wq = (const float*)d_in[1];
  const float* Wk = (const float*)d_in[2];
  const float* Wv = (const float*)d_in[3];
  const float* Wo = (const float*)d_in[4];
  const float* qs = (const float*)d_in[5];
  const float* ks = (const float*)d_in[6];
  float* out = (float*)d_out;

  const int MT  = B_ * T_;      // 4096 rows
  const int KVD = NKV * HD;     // 512

  char* w = (char*)d_ws;
  auto take = [&](size_t bytes) -> char* {
    char* p = w; w += (bytes + 255) & ~(size_t)255; return p;
  };
  __bf16* xb  = (__bf16*)take((size_t)MT * DM * 2);
  __bf16* wqb = (__bf16*)take((size_t)DM * DM * 2);
  __bf16* wkb = (__bf16*)take((size_t)KVD * DM * 2);
  __bf16* wvb = (__bf16*)take((size_t)KVD * DM * 2);
  __bf16* wob = (__bf16*)take((size_t)DM * DM * 2);
  float*  qf  = (float*)take((size_t)MT * DM * 4);
  float*  kf  = (float*)take((size_t)MT * KVD * 4);
  float*  vf  = (float*)take((size_t)MT * KVD * 4);
  __bf16* qb  = (__bf16*)take((size_t)MT * DM * 2);
  __bf16* kb  = (__bf16*)take((size_t)MT * KVD * 2);
  __bf16* vb  = (__bf16*)take((size_t)MT * KVD * 2);
  __bf16* aob = (__bf16*)take((size_t)MT * DM * 2);

  auto cvt = [&](const float* src, __bf16* dst, long n) {
    unsigned blocks = (unsigned)((n + 1023) / 1024);
    cvt_f32_bf16<<<blocks, 256, 0, stream>>>(src, dst, n);
  };
  cvt(x,  xb,  (long)MT * DM);
  cvt(Wq, wqb, (long)DM * DM);
  cvt(Wk, wkb, (long)KVD * DM);
  cvt(Wv, wvb, (long)KVD * DM);
  cvt(Wo, wob, (long)DM * DM);

  // Projections
  gemm_bf16_nt<<<(MT / 128) * (DM / 128), 256, 0, stream>>>(xb, wqb, qf, MT, DM, DM);
  gemm_bf16_nt<<<(MT / 128) * (KVD / 128), 256, 0, stream>>>(xb, wkb, kf, MT, KVD, DM);
  gemm_bf16_nt<<<(MT / 128) * (KVD / 128), 256, 0, stream>>>(xb, wvb, vf, MT, KVD, DM);

  // QK RMS norm (+scale) -> bf16 ; V -> bf16
  qknorm_kernel<<<(MT * NH) / 8, 256, 0, stream>>>(qf, qs, qb, MT * NH);
  qknorm_kernel<<<(MT * NKV) / 8, 256, 0, stream>>>(kf, ks, kb, MT * NKV);
  cvt(vf, vb, (long)MT * KVD);

  // Attention
  flash_attn_kernel<<<B_ * NH * (T_ / 128), 256, 0, stream>>>(qb, kb, vb, aob);

  // Output projection -> fp32 d_out
  gemm_bf16_nt<<<(MT / 128) * (DM / 128), 256, 0, stream>>>(aob, wob, out, MT, DM, DM);
}
